// MarketLevelFactorLearning_60713657697195
// MI455X (gfx1250) — matrix operand order repaired and checked by reading, not compiled, not run on hardware
//
#include <hip/hip_runtime.h>
#include <hip/hip_bf16.h>

#define Bc 4
#define Tc 64
#define Ic 512
#define FDc 128
#define Lc 5
#define OUTc 256

typedef __attribute__((ext_vector_type(16))) _Float16 v16h;
typedef __attribute__((ext_vector_type(8)))  float    v8f;
typedef __attribute__((ext_vector_type(4)))  float    f4;
typedef __attribute__((ext_vector_type(4)))  _Float16 h4;
typedef __attribute__((ext_vector_type(2)))  _Float16 h2;

// ---- workspace layout (float offsets) ----
static const size_t OFF_PROJ    = 0;          // 16,777,216 halves = 8,388,608 floats
static const size_t OFF_STOCK   = 8388608;    // 512*256
static const size_t OFF_SUMR    = 8519680;    // B*T*256 = 65536
static const size_t OFF_SUMRG   = 8585216;    // B*T*2*256 = 131072
static const size_t OFF_SUMETA  = 8716288;    // 256
static const size_t OFF_SUMETAG = 8716544;    // 512
static const size_t OFF_FN16    = 8717056;    // 512*256 halves = 65536 floats
static const size_t OFF_SIM     = 8782592;    // 512*512
static const size_t OFF_PEREX   = 9044736;    // 512
static const size_t OFF_GRP     = 9045248;    // 64*512 ints
static const size_t OFF_SCAL    = 9078016;    // 8

__global__ void k_zero(float* p, int n) {
  int i = blockIdx.x * 256 + threadIdx.x;
  if (i < n) p[i] = 0.0f;
}

// grp[t*I + i] = (position of i in perms[t] < I/2) ? 0 : 1
__global__ void k_rank(const int* __restrict__ perms, int* __restrict__ grp) {
  int t = blockIdx.x, j = threadIdx.x;
  int i = perms[t * Ic + j];
  grp[t * Ic + i] = (j < (Ic / 2)) ? 0 : 1;
}

// stock_proj[i, d] = sum_j stockID[i,j] * W_iota_w[d,j]
__global__ void k_stockproj(const float* __restrict__ stockID,
                            const float* __restrict__ Wio,
                            float* __restrict__ sp) {
  int idx = blockIdx.x * 256 + threadIdx.x;     // 512*256 threads
  int i = idx >> 8, d = idx & 255;
  const float* si = stockID + i * Ic;
  const float* wd = Wio + d * Ic;
  float acc = 0.0f;
  for (int j = 0; j < Ic; ++j) acc += si[j] * wd[j];
  sp[idx] = acc;
}

// proj = E(131072x128) @ W^T(128x128) + bias, output f16. WMMA f16->f32.
// W is converted to f16 once per block into LDS (32 KB); B-fragments come from
// ds_load, A-fragments stream from global and overlap with the WMMA pipe.
__global__ void __launch_bounds__(256) k_proj(const float* __restrict__ E,
                                              const float* __restrict__ W,
                                              const float* __restrict__ bias,
                                              _Float16* __restrict__ P) {
  __shared__ _Float16 Wl[FDc * FDc];  // 16384 halves = 32 KB

  // cooperative f32->f16 conversion of W into LDS (8192 packed pairs)
  {
    h2* wp = (h2*)Wl;
#pragma unroll
    for (int k = 0; k < 32; ++k) {
      int idx2 = threadIdx.x + 256 * k;
      h2 pk;
      pk[0] = (_Float16)W[2 * idx2];
      pk[1] = (_Float16)W[2 * idx2 + 1];
      wp[idx2] = pk;
    }
  }
  __syncthreads();

  int wave = threadIdx.x >> 5, lane = threadIdx.x & 31;
  int m = lane & 15, hi = lane >> 4;
  long row0 = (long)blockIdx.x * 128 + wave * 16;

  v8f acc[8];
#pragma unroll
  for (int nt = 0; nt < 8; ++nt)
#pragma unroll
    for (int v = 0; v < 8; ++v) acc[nt][v] = 0.0f;

  const float* arow = E + (row0 + m) * 128;
#pragma unroll
  for (int kt = 0; kt < 4; ++kt) {
    v16h a;
    const float* ap = arow + kt * 32 + hi * 16;
#pragma unroll
    for (int e = 0; e < 8; ++e) {
      a[e]     = (_Float16)ap[e];
      a[e + 8] = (_Float16)ap[8 + e];
    }
#pragma unroll
    for (int nt = 0; nt < 8; ++nt) {
      int ncol = nt * 16 + m;
      v16h bv = *(const v16h*)(Wl + ncol * FDc + kt * 32 + hi * 16);
      acc[nt] = __builtin_amdgcn_wmma_f32_16x16x32_f16(
          false, a, false, bv, (short)0, acc[nt], false, false);
    }
  }
#pragma unroll
  for (int nt = 0; nt < 8; ++nt) {
    int ncol = nt * 16 + m;
    float bb = bias[ncol];
#pragma unroll
    for (int v = 0; v < 8; ++v) {
      long r = row0 + v + 8 * hi;
      P[r * 128 + ncol] = (_Float16)(acc[nt][v] + bb);
    }
  }
}

// attention over L=5 window + eta + weighted-mean accumulation; one block per (b,t)
__global__ void __launch_bounds__(256) k_attn(const float* __restrict__ E,
                                              const _Float16* __restrict__ P,
                                              const float* __restrict__ SP,
                                              const float* __restrict__ w_eta,
                                              const int* __restrict__ grp,
                                              float* __restrict__ sum_r,
                                              float* __restrict__ sum_rg,
                                              float* __restrict__ sum_eta,
                                              float* __restrict__ sum_eta_g,
                                              float* __restrict__ out_rt) {
  int bt = blockIdx.x;
  int b = bt / Tc, t = bt % Tc;
  int wave = threadIdx.x >> 5, lane = threadIdx.x & 31;
  int d0 = lane * 4;

  f4 we0 = *(const f4*)(w_eta + d0);
  f4 we1 = *(const f4*)(w_eta + 128 + d0);

  float accr[8];
  float accg0[8], accg1[8];
#pragma unroll
  for (int j = 0; j < 8; ++j) { accr[j] = 0.f; accg0[j] = 0.f; accg1[j] = 0.f; }
  float acce = 0.f, aceg0 = 0.f, aceg1 = 0.f;
  const float iscl = 0.08838834764831845f;  // 1/sqrt(128)

  for (int i = wave; i < Ic; i += 8) {
    long base_q = ((long)(b * Tc + t) * Ic + i) * 128;

    // pull the next i-chunk toward the WGP while we do this one's math
    if (i + 8 < Ic) {
      __builtin_prefetch((const void*)(P + base_q + 8 * 128 + d0), 0, 1);
      __builtin_prefetch((const void*)(E + base_q + 8 * 128 + d0), 0, 1);
    }

    h4 qv = *(const h4*)(P + base_q + d0);
    float q0 = (float)qv[0], q1 = (float)qv[1], q2 = (float)qv[2], q3 = (float)qv[3];

    float sc[Lc];
    f4 ew[Lc];
#pragma unroll
    for (int l = 0; l < Lc; ++l) {
      int tl = t + l - (Lc - 1);
      int tlc = tl < 0 ? 0 : tl;
      long base_k = ((long)(b * Tc + tlc) * Ic + i) * 128;
      h4 kv = *(const h4*)(P + base_k + d0);
      float p = q0 * (float)kv[0] + q1 * (float)kv[1] + q2 * (float)kv[2] + q3 * (float)kv[3];
#pragma unroll
      for (int off = 16; off >= 1; off >>= 1) p += __shfl_xor(p, off, 32);
      sc[l] = p * iscl;
      ew[l] = *(const f4*)(E + base_k + d0);
    }
    float mx = -3.0e38f;
#pragma unroll
    for (int l = 0; l < Lc; ++l)
      if ((t + l - (Lc - 1)) >= 0 && sc[l] > mx) mx = sc[l];
    float w[Lc], sw = 0.f;
#pragma unroll
    for (int l = 0; l < Lc; ++l) {
      w[l] = ((t + l - (Lc - 1)) >= 0) ? __expf(sc[l] - mx) : 0.0f;
      sw += w[l];
    }
    float isw = 1.0f / sw;
    f4 rh;
#pragma unroll
    for (int j = 0; j < 4; ++j) {
      float acc = 0.f;
#pragma unroll
      for (int l = 0; l < Lc; ++l) acc += w[l] * ew[l][j];
      rh[j] = acc * isw;
    }
    f4 ec = ew[Lc - 1];  // e_seq[b,t,i,:] slice

    const float* sp = SP + i * 256 + d0;
    f4 s0 = *(const f4*)sp;
    f4 s1 = *(const f4*)(sp + 128);
    float ep = 0.f;
#pragma unroll
    for (int j = 0; j < 4; ++j)
      ep += (ec[j] + s0[j]) * we0[j] + (rh[j] + s1[j]) * we1[j];
#pragma unroll
    for (int off = 16; off >= 1; off >>= 1) ep += __shfl_xor(ep, off, 32);
    float eta = ep > 0.f ? ep : 0.f;

    if (t == Tc - 1) {
      float* rt = out_rt + ((long)b * Ic + i) * 256 + d0;
#pragma unroll
      for (int j = 0; j < 4; ++j) { rt[j] = ec[j]; rt[128 + j] = rh[j]; }
    }

    int g = grp[t * Ic + i];
    float e0 = (g == 0) ? eta : 0.0f;
    float e1 = eta - e0;
    acce += eta; aceg0 += e0; aceg1 += e1;
#pragma unroll
    for (int j = 0; j < 4; ++j) {
      accr[j]      += eta * ec[j];  accr[4 + j]  += eta * rh[j];
      accg0[j]     += e0 * ec[j];   accg0[4 + j] += e0 * rh[j];
      accg1[j]     += e1 * ec[j];   accg1[4 + j] += e1 * rh[j];
    }
  }

  long sb = (long)bt * 256;
  long gb = (long)bt * 512;
#pragma unroll
  for (int j = 0; j < 4; ++j) {
    atomicAdd(&sum_r[sb + d0 + j],        accr[j]);
    atomicAdd(&sum_r[sb + 128 + d0 + j],  accr[4 + j]);
    atomicAdd(&sum_rg[gb + d0 + j],             accg0[j]);
    atomicAdd(&sum_rg[gb + 128 + d0 + j],       accg0[4 + j]);
    atomicAdd(&sum_rg[gb + 256 + d0 + j],       accg1[j]);
    atomicAdd(&sum_rg[gb + 256 + 128 + d0 + j], accg1[4 + j]);
  }
  if (lane == 0) {
    atomicAdd(&sum_eta[bt], acce);
    atomicAdd(&sum_eta_g[bt * 2 + 0], aceg0);
    atomicAdd(&sum_eta_g[bt * 2 + 1], aceg1);
  }
}

__global__ void k_mt(const float* __restrict__ sum_r, const float* __restrict__ sum_eta,
                     float* __restrict__ out_mt) {
  int b = blockIdx.x, d = threadIdx.x;
  int bt = b * Tc + (Tc - 1);
  out_mt[b * 256 + d] = sum_r[bt * 256 + d] / (sum_eta[bt] + 1e-6f);
}

// build normalized f16 feature rows; row = g*256 + t*B + b
__global__ void k_feats(const float* __restrict__ sum_rg, const float* __restrict__ sum_eta_g,
                        _Float16* __restrict__ fn16) {
  int r = blockIdx.x, tid = threadIdx.x;  // 64 threads
  int g = r >> 8, rem = r & 255;
  int t = rem >> 2, b = rem & 3;
  int bt = b * Tc + t;
  const float* s = sum_rg + (long)bt * 512 + g * 256;
  float den = sum_eta_g[bt * 2 + g] + 1e-6f;
  float mv[4], p = 0.f;
#pragma unroll
  for (int k = 0; k < 4; ++k) { mv[k] = s[tid * 4 + k] / den; p += mv[k] * mv[k]; }
  __shared__ float red[64];
  red[tid] = p;
  __syncthreads();
  for (int st = 32; st > 0; st >>= 1) {
    if (tid < st) red[tid] += red[tid + st];
    __syncthreads();
  }
  float nrm = sqrtf(red[0]);
  nrm = nrm < 1e-12f ? 1e-12f : nrm;
  float inv = 1.0f / nrm;
#pragma unroll
  for (int k = 0; k < 4; ++k) fn16[r * 256 + tid * 4 + k] = (_Float16)(mv[k] * inv);
}

// sim = Fn(512x256) @ Fn^T, WMMA f16->f32; 16 blocks of 128x128
__global__ void __launch_bounds__(256) k_sim(const _Float16* __restrict__ F,
                                             float* __restrict__ sim) {
  int wave = threadIdx.x >> 5, lane = threadIdx.x & 31;
  int m = lane & 15, hi = lane >> 4;
  int tr = blockIdx.x >> 2, tc = blockIdx.x & 3;
  int row0 = tr * 128 + wave * 16;
  int col0 = tc * 128;

  v8f acc[8];
#pragma unroll
  for (int nt = 0; nt < 8; ++nt)
#pragma unroll
    for (int v = 0; v < 8; ++v) acc[nt][v] = 0.0f;

  const _Float16* arow = F + (row0 + m) * 256;
#pragma unroll
  for (int kt = 0; kt < 8; ++kt) {
    v16h a;
    const _Float16* ap = arow + kt * 32 + hi * 16;
#pragma unroll
    for (int e = 0; e < 8; ++e) { a[e] = ap[e]; a[e + 8] = ap[8 + e]; }
#pragma unroll
    for (int nt = 0; nt < 8; ++nt) {
      int ncol = col0 + nt * 16 + m;
      const _Float16* bp = F + ncol * 256 + kt * 32 + hi * 16;
      v16h bv;
#pragma unroll
      for (int e = 0; e < 16; ++e) bv[e] = bp[e];
      acc[nt] = __builtin_amdgcn_wmma_f32_16x16x32_f16(
          false, a, false, bv, (short)0, acc[nt], false, false);
    }
  }
#pragma unroll
  for (int nt = 0; nt < 8; ++nt)
#pragma unroll
    for (int v = 0; v < 8; ++v)
      sim[(row0 + v + 8 * hi) * 512 + col0 + nt * 16 + m] = acc[nt][v];
}

// per-row masked logsumexp contrastive term
__global__ void k_contrast(const float* __restrict__ sim, float* __restrict__ per_ex) {
  int r = blockIdx.x, tid = threadIdx.x;  // 64 threads
  int lr = (r & 255) >> 2;
  const float INV_TEMP = 1.0f / 0.07f;
  float v1[8], v2[8];
#pragma unroll
  for (int k = 0; k < 8; ++k) {
    int c = tid + k * 64;
    float s = sim[r * 512 + c] * INV_TEMP;
    int lc = (c & 255) >> 2;
    bool eq = (lc == lr), diag = (c == r);
    v1[k] = diag ? 0.0f : (eq ? s : -1000.0f);
    v2[k] = diag ? 0.0f : (eq ? -1000.0f : s);
  }
  float m1 = -3.0e38f, m2 = -3.0e38f;
#pragma unroll
  for (int k = 0; k < 8; ++k) { m1 = fmaxf(m1, v1[k]); m2 = fmaxf(m2, v2[k]); }
  __shared__ float red[64];
  red[tid] = m1; __syncthreads();
  for (int st = 32; st > 0; st >>= 1) { if (tid < st) red[tid] = fmaxf(red[tid], red[tid + st]); __syncthreads(); }
  m1 = red[0]; __syncthreads();
  red[tid] = m2; __syncthreads();
  for (int st = 32; st > 0; st >>= 1) { if (tid < st) red[tid] = fmaxf(red[tid], red[tid + st]); __syncthreads(); }
  m2 = red[0]; __syncthreads();
  float s1 = 0.f, s2 = 0.f;
#pragma unroll
  for (int k = 0; k < 8; ++k) { s1 += __expf(v1[k] - m1); s2 += __expf(v2[k] - m2); }
  red[tid] = s1; __syncthreads();
  for (int st = 32; st > 0; st >>= 1) { if (tid < st) red[tid] += red[tid + st]; __syncthreads(); }
  s1 = red[0]; __syncthreads();
  red[tid] = s2; __syncthreads();
  for (int st = 32; st > 0; st >>= 1) { if (tid < st) red[tid] += red[tid + st]; __syncthreads(); }
  s2 = red[0];
  if (tid == 0) per_ex[r] = (m1 + __logf(s1)) - (m2 + __logf(s2));
}

// sync label + 2-layer head + NLL accumulation; one block per (b, t<T-1)
__global__ void __launch_bounds__(256) k_sync(const float* __restrict__ E,
                                              const float* __restrict__ sum_r,
                                              const float* __restrict__ sum_eta,
                                              const float* __restrict__ w1,
                                              const float* __restrict__ b1,
                                              const float* __restrict__ w2,
                                              const float* __restrict__ b2,
                                              float* __restrict__ acc) {
  int blk = blockIdx.x;
  int b = blk / (Tc - 1), t = blk % (Tc - 1);
  int tid = threadIdx.x;
  __shared__ float sm[256];
  __shared__ float sh[256];
  __shared__ float red[256];
  __shared__ int cnt[2];
  if (tid < 2) cnt[tid] = 0;
  __syncthreads();

  int pc = 0, nc = 0;
  for (int i = tid; i < Ic; i += 256) {
    float c0 = E[(((long)(b * Tc + t) * Ic) + i) * 128 + 3];
    float c1 = E[(((long)(b * Tc + t + 1) * Ic) + i) * 128 + 3];
    float ret = __logf(c1 / (c0 + 1e-8f));
    pc += (ret > 0.0f) ? 1 : 0;
    nc += (ret < 0.0f) ? 1 : 0;
  }
  atomicAdd(&cnt[0], pc);
  atomicAdd(&cnt[1], nc);

  int bt = b * Tc + t;
  sm[tid] = sum_r[bt * 256 + tid] / (sum_eta[bt] + 1e-6f);
  __syncthreads();

  float h = b1[tid];
  const float* wr = w1 + tid * 256;
  for (int d = 0; d < 256; ++d) h += wr[d] * sm[d];
  sh[tid] = h > 0.f ? h : 0.f;
  __syncthreads();

  float lg[3];
  for (int j = 0; j < 3; ++j) {
    red[tid] = w2[j * 256 + tid] * sh[tid];
    __syncthreads();
    for (int st = 128; st > 0; st >>= 1) {
      if (tid < st) red[tid] += red[tid + st];
      __syncthreads();
    }
    lg[j] = red[0] + b2[j];
    __syncthreads();
  }
  if (tid == 0) {
    float pr = cnt[0] * (1.0f / Ic), nr = cnt[1] * (1.0f / Ic);
    int lbl = (pr >= 0.6f) ? 0 : ((nr >= 0.6f) ? 1 : 2);
    float mx = fmaxf(lg[0], fmaxf(lg[1], lg[2]));
    float lse = mx + __logf(__expf(lg[0] - mx) + __expf(lg[1] - mx) + __expf(lg[2] - mx));
    atomicAdd(acc, -(lg[lbl] - lse));
  }
}

__global__ void k_final(const float* __restrict__ per_ex, const float* __restrict__ sync_acc,
                        float* __restrict__ out_losses) {
  __shared__ float red[256];
  int tid = threadIdx.x;
  red[tid] = per_ex[tid] + per_ex[tid + 256];
  __syncthreads();
  for (int st = 128; st > 0; st >>= 1) {
    if (tid < st) red[tid] += red[tid + st];
    __syncthreads();
  }
  if (tid == 0) {
    float lc = -red[0] / 512.0f;
    float ls = sync_acc[0] / (float)(Bc * (Tc - 1));
    out_losses[0] = lc + ls;
    out_losses[1] = lc;
    out_losses[2] = ls;
  }
}

extern "C" void kernel_launch(void* const* d_in, const int* in_sizes, int n_in,
                              void* d_out, int out_size, void* d_ws, size_t ws_size,
                              hipStream_t stream) {
  const float* e_seq   = (const float*)d_in[0];
  const float* stockID = (const float*)d_in[1];
  const float* W_s_w   = (const float*)d_in[2];
  const float* W_s_b   = (const float*)d_in[3];
  const float* W_iota  = (const float*)d_in[4];
  const float* w_eta   = (const float*)d_in[5];
  const float* sw1     = (const float*)d_in[6];
  const float* sb1     = (const float*)d_in[7];
  const float* sw2     = (const float*)d_in[8];
  const float* sb2     = (const float*)d_in[9];
  const int*   perms   = (const int*)d_in[10];

  float* ws = (float*)d_ws;
  _Float16* proj16   = (_Float16*)(ws + OFF_PROJ);
  float*    stock    = ws + OFF_STOCK;
  float*    sum_r    = ws + OFF_SUMR;
  float*    sum_rg   = ws + OFF_SUMRG;
  float*    sum_eta  = ws + OFF_SUMETA;
  float*    sum_etag = ws + OFF_SUMETAG;
  _Float16* fn16     = (_Float16*)(ws + OFF_FN16);
  float*    sim      = ws + OFF_SIM;
  float*    per_ex   = ws + OFF_PEREX;
  int*      grp      = (int*)(ws + OFF_GRP);
  float*    scal     = ws + OFF_SCAL;

  float* out        = (float*)d_out;
  float* out_rt     = out;                          // B*I*256
  float* out_mt     = out + (size_t)Bc * Ic * 256;  // B*256
  float* out_losses = out_mt + Bc * 256;            // 3 scalars

  // zero accumulators (SUMR..SUMETAG are contiguous) + loss scalar
  int nz = 65536 + 131072 + 256 + 512;
  k_zero<<<(nz + 255) / 256, 256, 0, stream>>>(ws + OFF_SUMR, nz);
  k_zero<<<1, 256, 0, stream>>>(scal, 8);

  k_rank<<<Tc, Ic, 0, stream>>>(perms, grp);
  k_stockproj<<<(Ic * OUTc) / 256, 256, 0, stream>>>(stockID, W_iota, stock);
  k_proj<<<(Bc * Tc * Ic) / 128, 256, 0, stream>>>(e_seq, W_s_w, W_s_b, proj16);
  k_attn<<<Bc * Tc, 256, 0, stream>>>(e_seq, proj16, stock, w_eta, grp,
                                      sum_r, sum_rg, sum_eta, sum_etag, out_rt);
  k_mt<<<Bc, 256, 0, stream>>>(sum_r, sum_eta, out_mt);
  k_feats<<<512, 64, 0, stream>>>(sum_rg, sum_etag, fn16);
  k_sim<<<16, 256, 0, stream>>>(fn16, sim);
  k_contrast<<<512, 64, 0, stream>>>(sim, per_ex);
  k_sync<<<Bc * (Tc - 1), 256, 0, stream>>>(e_seq, sum_r, sum_eta, sw1, sb1, sw2, sb2, scal);
  k_final<<<1, 256, 0, stream>>>(per_ex, scal, out_losses);
}
